// deform_conv2d_wrapper_62285615727270
// MI455X (gfx1250) — matrix operand order repaired and checked by reading, not compiled, not run on hardware
//
#include <hip/hip_runtime.h>

typedef __attribute__((ext_vector_type(16))) _Float16 v16h;
typedef __attribute__((ext_vector_type(8)))  _Float16 v8h;
typedef __attribute__((ext_vector_type(8)))  float    v8f;

#define Ncfg   8
#define Ccfg   256
#define Hcfg   64
#define Wcfg   64
#define HWcfg  4096            // 64*64
#define CHW    1048576         // 256*4096
#define OFFCH  73728           // 18*4096
#define KTAPS  9
#define LDSK   40              // 32 halves + 8 pad per B row

// ---------------------------------------------------------------------------
// One-time weight repack: fp32 [cout][c][kh][kw] -> f16 [k][cout][c]
// ---------------------------------------------------------------------------
__global__ void repack_weights_k(const float* __restrict__ w, _Float16* __restrict__ wr) {
    int idx = blockIdx.x * 256 + threadIdx.x;          // 9*256*256 = 589824
    if (idx >= KTAPS * Ccfg * Ccfg) return;
    int k    = idx >> 16;
    int r    = idx & 65535;
    int cout = r >> 8;
    int c    = r & 255;
    wr[idx] = (_Float16)w[cout * 2304 + c * 9 + k];
}

// ---------------------------------------------------------------------------
// Fused deformable-conv implicit GEMM.
// grid = (P/128 = 256, Cout/128 = 2), block = 256 (8 waves)
// ---------------------------------------------------------------------------
__global__ __launch_bounds__(256) void deform_gemm_k(
    const float* __restrict__ x, const float* __restrict__ off,
    const _Float16* __restrict__ wr, const float* __restrict__ bias,
    float* __restrict__ out)
{
    __shared__ __align__(16) _Float16 ldsB[128 * LDSK];

    const int tid    = threadIdx.x;
    const int lane   = tid & 31;
    const int wave   = tid >> 5;        // 0..7
    const int lane16 = lane & 15;
    const int laneHi = lane >> 4;       // 0/1
    const int mw     = wave >> 1;       // 0..3 (Cout dir)
    const int nw     = wave & 1;        // 0..1 (P dir)

    const int coutBase = blockIdx.y * 128;
    const int pBase    = blockIdx.x * 128;
    const int n        = pBase >> 12;     // image (128 | 4096 -> never straddles)
    const int sBase    = pBase & 4095;    // spatial base within image

    // --- sampling-thread mapping: 2 threads per position, 16 channels each ---
    const int pl    = tid & 127;          // position within tile
    const int cHalf = tid >> 7;           // 0/1 -> channel half of 32-chunk
    const int s     = sBase + pl;
    const int ho    = s >> 6;
    const int wo    = s & 63;
    const float* __restrict__ xn   = x   + n * CHW;
    const float* __restrict__ offn = off + n * OFFCH + s;

    v8f acc[2][4];
    const v8f vzero = {0.f, 0.f, 0.f, 0.f, 0.f, 0.f, 0.f, 0.f};
#pragma unroll
    for (int i = 0; i < 2; ++i)
#pragma unroll
        for (int j = 0; j < 4; ++j) acc[i][j] = vzero;

    for (int k = 0; k < KTAPS; ++k) {
        const int kh = k / 3;
        const int kw = k - 3 * kh;

        // ---- bilinear sampling parameters, once per (tap, position) ----
        const float offy = offn[(2 * k) * HWcfg];
        const float offx = offn[(2 * k + 1) * HWcfg];
        const float py = (float)(ho - 1 + kh) + offy;
        const float px = (float)(wo - 1 + kw) + offx;
        const float y0f = floorf(py), x0f = floorf(px);
        const float ly = py - y0f, lx = px - x0f;
        const float hy = 1.f - ly, hx = 1.f - lx;
        const int y0 = (int)y0f, x0 = (int)x0f;
        const int y1 = y0 + 1,  x1 = x0 + 1;
        const float vy0 = (y0 >= 0 && y0 < Hcfg) ? 1.f : 0.f;
        const float vy1 = (y1 >= 0 && y1 < Hcfg) ? 1.f : 0.f;
        const float vx0 = (x0 >= 0 && x0 < Wcfg) ? 1.f : 0.f;
        const float vx1 = (x1 >= 0 && x1 < Wcfg) ? 1.f : 0.f;
        const float w00 = hy * hx * vy0 * vx0;
        const float w01 = hy * lx * vy0 * vx1;
        const float w10 = ly * hx * vy1 * vx0;
        const float w11 = ly * lx * vy1 * vx1;
        const int yc0 = y0 < 0 ? 0 : (y0 > 63 ? 63 : y0);
        const int yc1 = y1 < 0 ? 0 : (y1 > 63 ? 63 : y1);
        const int xc0 = x0 < 0 ? 0 : (x0 > 63 ? 63 : x0);
        const int xc1 = x1 < 0 ? 0 : (x1 > 63 ? 63 : x1);
        const int i00 = yc0 * Wcfg + xc0;
        const int i01 = yc0 * Wcfg + xc1;
        const int i10 = yc1 * Wcfg + xc0;
        const int i11 = yc1 * Wcfg + xc1;

        const _Float16* __restrict__ wk = wr + k * (Ccfg * Ccfg);   // [cout][c]

        for (int c0 = 0; c0 < Ccfg; c0 += 32) {
            // ---- build B tile: cols[c0..c0+31][pTile] as LDS [p][c] f16 ----
            const float* __restrict__ xb = xn + (c0 + cHalf * 16) * HWcfg;
            _Float16* __restrict__ row = ldsB + pl * LDSK + cHalf * 16;
#pragma unroll
            for (int cc = 0; cc < 16; cc += 2) {
                const float* ch0 = xb + cc * HWcfg;
                const float* ch1 = ch0 + HWcfg;
                float v0 = w00 * ch0[i00] + w01 * ch0[i01] + w10 * ch0[i10] + w11 * ch0[i11];
                float v1 = w00 * ch1[i00] + w01 * ch1[i01] + w10 * ch1[i10] + w11 * ch1[i11];
                union { _Float16 h[2]; unsigned u; } pk;
                pk.h[0] = (_Float16)v0;
                pk.h[1] = (_Float16)v1;
                *(unsigned*)(row + cc) = pk.u;      // 4B-aligned packed store
            }
            __syncthreads();

            // ---- A fragments straight from repacked global weights ----
            v16h afrag[2];
#pragma unroll
            for (int mt = 0; mt < 2; ++mt) {
                const _Float16* ap = wk + (coutBase + mw * 32 + mt * 16 + lane16) * Ccfg + c0;
                union { v16h v; v8h h2[2]; } ua;
                ua.h2[0] = *(const v8h*)(ap + laneHi * 8);        // K 0-7 / 8-15
                ua.h2[1] = *(const v8h*)(ap + 16 + laneHi * 8);   // K 16-23 / 24-31
                afrag[mt] = ua.v;
            }

            // ---- B fragments from LDS + WMMA ----
#pragma unroll
            for (int nt = 0; nt < 4; ++nt) {
                const _Float16* bp = ldsB + (nw * 64 + nt * 16 + lane16) * LDSK;
                union { v16h v; v8h h2[2]; } ub;
                ub.h2[0] = *(const v8h*)(bp + laneHi * 8);        // K 0-7  / 8-15
                ub.h2[1] = *(const v8h*)(bp + 16 + laneHi * 8);   // K 16-23 / 24-31
                const v16h bfrag = ub.v;
#pragma unroll
                for (int mt = 0; mt < 2; ++mt)
                    acc[mt][nt] = __builtin_amdgcn_wmma_f32_16x16x32_f16(
                        false, afrag[mt], false, bfrag,
                        (short)0, acc[mt][nt], false, false);
            }
            __syncthreads();
        }
    }

    // ---- epilogue: add bias, write f32 NCHW ----
    float* __restrict__ outBase = out + n * CHW + sBase + nw * 64 + lane16;
#pragma unroll
    for (int mt = 0; mt < 2; ++mt) {
#pragma unroll
        for (int r = 0; r < 8; ++r) {
            const int m = coutBase + mw * 32 + mt * 16 + laneHi * 8 + r;
            const float b = bias[m];
#pragma unroll
            for (int nt = 0; nt < 4; ++nt)
                outBase[m * HWcfg + nt * 16] = acc[mt][nt][r] + b;
        }
    }
}

// ---------------------------------------------------------------------------
extern "C" void kernel_launch(void* const* d_in, const int* in_sizes, int n_in,
                              void* d_out, int out_size, void* d_ws, size_t ws_size,
                              hipStream_t stream) {
    const float* x    = (const float*)d_in[0];
    const float* off  = (const float*)d_in[1];
    const float* w    = (const float*)d_in[2];
    const float* bias = (const float*)d_in[3];
    float* out        = (float*)d_out;
    _Float16* wr      = (_Float16*)d_ws;     // 9*256*256 f16 = 1.18 MB scratch

    repack_weights_k<<<(KTAPS * Ccfg * Ccfg + 255) / 256, 256, 0, stream>>>(w, wr);

    dim3 grid(256, 2);                        // P/128, Cout/128
    deform_gemm_k<<<grid, 256, 0, stream>>>(x, off, wr, bias, out);
}
